// ParticleNetBlock_25039659336453
// MI455X (gfx1250) — matrix operand order repaired and checked by reading, not compiled, not run on hardware
//
#include <hip/hip_runtime.h>

// ---------------------------------------------------------------------------
// ParticleNet EdgeConv block for gfx1250 (MI455X), wave32 + WMMA bf16.
//   E = N*K edges, K = 16 (one node's edge set == one 16-row WMMA tile).
//   BN is training-mode over edges -> two global-stats passes; GEMMs are
//   recomputed instead of materialized ([E,64] f32 = 205MB each) because
//   x (12.8MB) is L2-resident and matrix-core FLOPs are essentially free.
//   b1/b2 cancel inside BatchNorm; b3 commutes with the per-node max.
// ---------------------------------------------------------------------------

typedef __bf16 bf16_t;
typedef __attribute__((ext_vector_type(16))) __bf16        v16bf;
typedef __attribute__((ext_vector_type(2)))  __bf16        v2bf;
typedef __attribute__((ext_vector_type(8)))  float         v8f;
typedef __attribute__((ext_vector_type(4)))  unsigned int  u32x4;

union FragU { u32x4 q[2]; v16bf v; };

#define LRELU_SLOPE 0.01f
#define BN_EPS_F 1e-5f
#define HSTRIDE 136   // 16x128 H tile row stride (halfs), padded: 272B/row -> 4-bank step, conflict free
#define ZSTRIDE 72    // 16x64 z tile row stride (halfs), padded: 144B/row -> 36-bank step, conflict free

// ---- WMMA GEMM: D[16x64] += A[16xK] * W[Kx64], A/W staged in LDS (bf16) ----
// A layout (16-bit 16x32): lane l -> row m = l&15, khalf = (l>>4)*8,
//   chunk0 = k0+khalf..+7, chunk1 = k0+16+khalf..+7   (two b128 loads)
// B layout (16-bit 32x16): lane l -> col n = l&15, khalf = (l>>4)*8,
//   chunk0 = k0+khalf..+7, chunk1 = k0+16+khalf..+7   (two b128 loads)
// W stored transposed in LDS as [n][k] so B chunks are contiguous in k.
template <int K, int ASTRIDE, int WSTRIDE>
__device__ __forceinline__ void gemm_bf16(const bf16_t* Ab, const bf16_t* Wb,
                                          v8f* acc, int lane) {
  const int mn = lane & 15;
  const int hi = lane >> 4;
  const int ka = hi * 8;
  const int kb = hi * 16;
  const bf16_t* arow = Ab + mn * ASTRIDE;
#pragma unroll
  for (int k0 = 0; k0 < K; k0 += 32) {
    FragU fa;
    fa.q[0] = *(const u32x4*)(arow + k0 + ka);
    fa.q[1] = *(const u32x4*)(arow + k0 + ka + 16);
#pragma unroll
    for (int nt = 0; nt < 4; ++nt) {
      const bf16_t* wrow = Wb + (nt * 16 + mn) * WSTRIDE;
      FragU fb;
      fb.q[0] = *(const u32x4*)(wrow + k0 + ka);
      fb.q[1] = *(const u32x4*)(wrow + k0 + ka + 16);
      acc[nt] = __builtin_amdgcn_wmma_f32_16x16x32_bf16(
          false, fa.v, false, fb.v, (short)0, acc[nt], false, false);
    }
  }
}

// ---- stage W[K][64] (f32, global) -> LDS transposed [n][K] bf16 ----
template <int K, int STRIDE>
__device__ __forceinline__ void stage_w(const float* __restrict__ Wg,
                                        bf16_t* Wl, int tid) {
  for (int idx = tid; idx < K * 64; idx += 128) {
    int k = idx >> 6, n = idx & 63;
    Wl[n * STRIDE + k] = (bf16_t)Wg[idx];
  }
}

// ---- build H[16][128] bf16 for node n: [x_i | x_j - x_i] ----
__device__ __forceinline__ void build_H(const float* __restrict__ x,
                                        const int* __restrict__ src, int n,
                                        bf16_t* hb, int lane) {
  const float2* x2 = (const float2*)x;  // row stride = 32 float2
  float2 xi = x2[(long)n * 32 + lane];  // cols 2*lane, 2*lane+1
  v2bf xib;
  xib.x = (bf16_t)xi.x;
  xib.y = (bf16_t)xi.y;
#pragma unroll
  for (int k = 0; k < 16; ++k)
    *(v2bf*)(hb + k * HSTRIDE + 2 * lane) = xib;
  const int* sp = src + n * 16;
#pragma unroll
  for (int k = 0; k < 16; ++k) {
    int s = sp[k];
    float2 xj = x2[(long)s * 32 + lane];  // L2-resident gather (x = 12.8MB)
    v2bf d;
    d.x = (bf16_t)(xj.x - xi.x);
    d.y = (bf16_t)(xj.y - xi.y);
    *(v2bf*)(hb + k * HSTRIDE + 64 + 2 * lane) = d;
  }
  asm volatile("" ::: "memory");  // order LDS writes vs. fragment reads
}

// ---- BN(scale/shift) + leaky-relu, D frags -> LDS z[16][ZSTRIDE] bf16 ----
// D layout: lane l, vgpr j -> row M = j + 8*(l>>4), col N = l&15 (+16*nt)
__device__ __forceinline__ void bn_lrelu_store(const v8f* acc, const float* af,
                                               const float* cf, bf16_t* zb,
                                               int lane) {
  const int hi = lane >> 4;
  const int nl = lane & 15;
#pragma unroll
  for (int nt = 0; nt < 4; ++nt) {
    int c = nt * 16 + nl;
    float a = af[c], sh = cf[c];
#pragma unroll
    for (int j = 0; j < 8; ++j) {
      float v = acc[nt][j] * a + sh;
      v = v > 0.f ? v : LRELU_SLOPE * v;
      zb[(hi * 8 + j) * ZSTRIDE + c] = (bf16_t)v;
    }
  }
  asm volatile("" ::: "memory");
}

// ---- per-column sum / sum-of-squares of a 16-row tile into registers ----
__device__ __forceinline__ void accum_stats(const v8f* acc, float* sAcc,
                                            float* qAcc) {
#pragma unroll
  for (int nt = 0; nt < 4; ++nt) {
    float s = 0.f, q = 0.f;
#pragma unroll
    for (int j = 0; j < 8; ++j) {
      float v = acc[nt][j];
      s += v;
      q += v * v;
    }
    s += __shfl_xor(s, 16, 32);
    q += __shfl_xor(q, 16, 32);
    sAcc[nt] += s;
    qAcc[nt] += q;
  }
}

__device__ __forceinline__ void block_reduce_stats(float* sbuf, float* qbuf,
                                                   const float* sAcc,
                                                   const float* qAcc,
                                                   float* gsum, float* gsq,
                                                   int tid, int lane) {
  if (lane < 16) {
#pragma unroll
    for (int nt = 0; nt < 4; ++nt) {
      int c = nt * 16 + lane;
      atomicAdd(&sbuf[c], sAcc[nt]);
      atomicAdd(&qbuf[c], qAcc[nt]);
    }
  }
  __syncthreads();
  if (tid < 64) {
    atomicAdd(&gsum[tid], sbuf[tid]);
    atomicAdd(&gsq[tid], qbuf[tid]);
  }
}

__device__ __forceinline__ void bn_coeffs(const float* stats_sum,
                                          const float* stats_sq,
                                          const float* g, const float* be,
                                          float Ef, float* af, float* cf,
                                          int tid) {
  if (tid < 64) {
    float m = stats_sum[tid] / Ef;
    float var = stats_sq[tid] / Ef - m * m;
    float a = g[tid] * rsqrtf(var + BN_EPS_F);
    af[tid] = a;
    cf[tid] = be[tid] - m * a;
  }
}

// ============================ kernels ======================================

__global__ void __launch_bounds__(256) k_init(float* __restrict__ stats) {
  if (threadIdx.x < 256) stats[threadIdx.x] = 0.f;
}

struct __align__(16) SharedS1 {
  bf16_t w1[64 * HSTRIDE];
  float sbuf[64], qbuf[64];
  bf16_t hbuf[4][16 * HSTRIDE];
};

__global__ void __launch_bounds__(128) k_stats1(
    const float* __restrict__ x, const int* __restrict__ src,
    const float* __restrict__ W1g, float* __restrict__ stats, int nNodes) {
  __shared__ SharedS1 sm;
  const int tid = threadIdx.x;
  stage_w<128, HSTRIDE>(W1g, sm.w1, tid);
  if (tid < 64) { sm.sbuf[tid] = 0.f; sm.qbuf[tid] = 0.f; }
  __syncthreads();
  const int lane = tid & 31, wave = tid >> 5;
  bf16_t* hb = sm.hbuf[wave];
  const int gw = blockIdx.x * 4 + wave, wstride = gridDim.x * 4;
  float sAcc[4] = {0, 0, 0, 0}, qAcc[4] = {0, 0, 0, 0};
  for (int n = gw; n < nNodes; n += wstride) {
    build_H(x, src, n, hb, lane);
    v8f acc[4] = {};
    gemm_bf16<128, HSTRIDE, HSTRIDE>(hb, sm.w1, acc, lane);
    accum_stats(acc, sAcc, qAcc);
  }
  block_reduce_stats(sm.sbuf, sm.qbuf, sAcc, qAcc, stats, stats + 64, tid, lane);
}

struct __align__(16) SharedS2 {
  bf16_t w1[64 * HSTRIDE];
  bf16_t w2[64 * ZSTRIDE];
  float af1[64], cf1[64];
  float sbuf[64], qbuf[64];
  bf16_t hbuf[4][16 * HSTRIDE];
};

__global__ void __launch_bounds__(128) k_stats2(
    const float* __restrict__ x, const int* __restrict__ src,
    const float* __restrict__ W1g, const float* __restrict__ W2g,
    const float* __restrict__ g1, const float* __restrict__ be1,
    float* __restrict__ stats, int nNodes, int nE) {
  __shared__ SharedS2 sm;
  const int tid = threadIdx.x;
  stage_w<128, HSTRIDE>(W1g, sm.w1, tid);
  stage_w<64, ZSTRIDE>(W2g, sm.w2, tid);
  bn_coeffs(stats, stats + 64, g1, be1, (float)nE, sm.af1, sm.cf1, tid);
  if (tid < 64) { sm.sbuf[tid] = 0.f; sm.qbuf[tid] = 0.f; }
  __syncthreads();
  const int lane = tid & 31, wave = tid >> 5;
  bf16_t* hb = sm.hbuf[wave];
  const int gw = blockIdx.x * 4 + wave, wstride = gridDim.x * 4;
  float sAcc[4] = {0, 0, 0, 0}, qAcc[4] = {0, 0, 0, 0};
  for (int n = gw; n < nNodes; n += wstride) {
    build_H(x, src, n, hb, lane);
    v8f a1[4] = {};
    gemm_bf16<128, HSTRIDE, HSTRIDE>(hb, sm.w1, a1, lane);
    bn_lrelu_store(a1, sm.af1, sm.cf1, hb, lane);  // reuse hb as z1[16][ZSTRIDE]
    v8f a2[4] = {};
    gemm_bf16<64, ZSTRIDE, ZSTRIDE>(hb, sm.w2, a2, lane);
    accum_stats(a2, sAcc, qAcc);
  }
  block_reduce_stats(sm.sbuf, sm.qbuf, sAcc, qAcc, stats + 128, stats + 192,
                     tid, lane);
}

struct __align__(16) SharedS3 {
  bf16_t w1[64 * HSTRIDE];
  bf16_t w2[64 * ZSTRIDE];
  bf16_t w3[64 * ZSTRIDE];
  float af1[64], cf1[64], af2[64], cf2[64];
  bf16_t hbuf[4][16 * HSTRIDE];
};

__global__ void __launch_bounds__(128) k_final(
    const float* __restrict__ x, const int* __restrict__ src,
    const float* __restrict__ W1g, const float* __restrict__ W2g,
    const float* __restrict__ W3g, const float* __restrict__ g1,
    const float* __restrict__ be1, const float* __restrict__ g2,
    const float* __restrict__ be2, const float* __restrict__ b3,
    const float* __restrict__ stats, float* __restrict__ out, int nNodes,
    int nE) {
  __shared__ SharedS3 sm;
  const int tid = threadIdx.x;
  stage_w<128, HSTRIDE>(W1g, sm.w1, tid);
  stage_w<64, ZSTRIDE>(W2g, sm.w2, tid);
  stage_w<64, ZSTRIDE>(W3g, sm.w3, tid);
  bn_coeffs(stats, stats + 64, g1, be1, (float)nE, sm.af1, sm.cf1, tid);
  bn_coeffs(stats + 128, stats + 192, g2, be2, (float)nE, sm.af2, sm.cf2, tid);
  __syncthreads();
  const int lane = tid & 31, wave = tid >> 5;
  bf16_t* hb = sm.hbuf[wave];
  const int gw = blockIdx.x * 4 + wave, wstride = gridDim.x * 4;
  for (int n = gw; n < nNodes; n += wstride) {
    build_H(x, src, n, hb, lane);
    v8f a1[4] = {};
    gemm_bf16<128, HSTRIDE, HSTRIDE>(hb, sm.w1, a1, lane);
    bn_lrelu_store(a1, sm.af1, sm.cf1, hb, lane);  // z1 in hb
    v8f a2[4] = {};
    gemm_bf16<64, ZSTRIDE, ZSTRIDE>(hb, sm.w2, a2, lane);
    bn_lrelu_store(a2, sm.af2, sm.cf2, hb, lane);  // z2 in hb
    v8f a3[4] = {};
    gemm_bf16<64, ZSTRIDE, ZSTRIDE>(hb, sm.w3, a3, lane);
    // segment-max over the node's 16 edges == column max of the 16x16 tiles
    // (dst = repeat(arange(N), K) groups each node's edges into one M-tile);
    // b3 is a per-column constant -> add after the max. Then residual + lrelu.
#pragma unroll
    for (int nt = 0; nt < 4; ++nt) {
      float v = a3[nt][0];
#pragma unroll
      for (int j = 1; j < 8; ++j) v = fmaxf(v, a3[nt][j]);
      v = fmaxf(v, __shfl_xor(v, 16, 32));
      if (lane < 16) {
        int c = nt * 16 + lane;
        float o = v + b3[c] + x[(long)n * 64 + c];
        out[(long)n * 64 + c] = o > 0.f ? o : LRELU_SLOPE * o;
      }
    }
  }
}

// ============================ launcher =====================================

extern "C" void kernel_launch(void* const* d_in, const int* in_sizes, int n_in,
                              void* d_out, int out_size, void* d_ws,
                              size_t ws_size, hipStream_t stream) {
  const float* x   = (const float*)d_in[0];
  const int*   ei  = (const int*)d_in[1];  // [2,E] int32: src row then dst row
  const float* W1  = (const float*)d_in[2];
  // d_in[3] = b1 : absorbed by BN1
  const float* g1  = (const float*)d_in[4];
  const float* be1 = (const float*)d_in[5];
  const float* W2  = (const float*)d_in[6];
  // d_in[7] = b2 : absorbed by BN2
  const float* g2  = (const float*)d_in[8];
  const float* be2 = (const float*)d_in[9];
  const float* W3  = (const float*)d_in[10];
  const float* b3  = (const float*)d_in[11];

  const int nNodes = in_sizes[0] / 64;
  const int nE     = in_sizes[1] / 2;
  const int* src   = ei;  // dst row unused: dst[e] == e / K by construction

  float* out   = (float*)d_out;
  float* stats = (float*)d_ws;  // 256 f32: sum1|sq1|sum2|sq2

  int blocks = (nNodes + 4 * 10 - 1) / (4 * 10);  // 4 waves/blk, ~10 nodes/wave
  if (blocks < 1) blocks = 1;

  k_init<<<1, 256, 0, stream>>>(stats);
  k_stats1<<<blocks, 128, 0, stream>>>(x, src, W1, stats, nNodes);
  k_stats2<<<blocks, 128, 0, stream>>>(x, src, W1, W2, g1, be1, stats, nNodes,
                                       nE);
  k_final<<<blocks, 128, 0, stream>>>(x, src, W1, W2, W3, g1, be1, g2, be2, b3,
                                      stats, out, nNodes, nE);
}
